// SparseEncoderResNet_13030930776772
// MI455X (gfx1250) — matrix operand order repaired and checked by reading, not compiled, not run on hardware
//
#include <hip/hip_runtime.h>
#include <stdint.h>

typedef __attribute__((ext_vector_type(16))) _Float16 v16h;
typedef __attribute__((ext_vector_type(8)))  _Float16 v8h;
typedef __attribute__((ext_vector_type(8)))  float    v8f;

union BU { v16h v; v8h h[2]; };

#define HD 192
#define EPSV 1e-5f
#define TPP 5  // taps staged per LDS phase (per Cout tile): 5+4 for 3x3, 1 for 1x1

// ---------------------------------------------------------------- utilities
__global__ void zero_f32(float* p, int n) {
  int i = blockIdx.x * blockDim.x + threadIdx.x;
  if (i < n) p[i] = 0.f;
}

// m0[b,y,x] = 1 - mask[b, (y/16)*16 + x/16]; also count active sites
__global__ void make_mask0(const int* __restrict__ mask, float* __restrict__ m0,
                           float* __restrict__ cnt) {
  __shared__ float red[256];
  int idx = blockIdx.x * 256 + threadIdx.x;  // B*256*256 total
  float keep = 0.f;
  if (idx < 8 * 65536) {
    int b = idx >> 16, off = idx & 65535, y = off >> 8, x = off & 255;
    keep = 1.f - (float)mask[b * 256 + (y >> 4) * 16 + (x >> 4)];
    m0[idx] = keep;
  }
  red[threadIdx.x] = keep;
  __syncthreads();
  for (int s = 128; s > 0; s >>= 1) {
    if (threadIdx.x < s) red[threadIdx.x] += red[threadIdx.x + s];
    __syncthreads();
  }
  if (threadIdx.x == 0) atomicAdd(cnt, red[0]);
}

// 3x3 stride-2 pad-1 max-pool of mask (active-output sites), + count
__global__ void down_mask(const float* __restrict__ mi, float* __restrict__ mo,
                          float* cnt, int B, int Hi, int Ho) {
  __shared__ float red[256];
  int idx = blockIdx.x * 256 + threadIdx.x;
  int total = B * Ho * Ho;
  float v = 0.f;
  if (idx < total) {
    int b = idx / (Ho * Ho); int r = idx - b * Ho * Ho;
    int y = r / Ho; int x = r - y * Ho;
    for (int ky = 0; ky < 3; ky++) {
      int iy = 2 * y - 1 + ky; if (iy < 0 || iy >= Hi) continue;
      for (int kx = 0; kx < 3; kx++) {
        int ix = 2 * x - 1 + kx; if (ix < 0 || ix >= Hi) continue;
        v = fmaxf(v, mi[((size_t)b * Hi + iy) * Hi + ix]);
      }
    }
    mo[idx] = v;
  }
  red[threadIdx.x] = v;
  __syncthreads();
  for (int s = 128; s > 0; s >>= 1) {
    if (threadIdx.x < s) red[threadIdx.x] += red[threadIdx.x + s];
    __syncthreads();
  }
  if (threadIdx.x == 0) atomicAdd(cnt, red[0]);
}

// ---------------------------------------------------------------- stem (1x1, C=3)
__global__ void __launch_bounds__(192)
stem_stats(const float* __restrict__ img, const float* __restrict__ m0,
           const float* __restrict__ w, float* __restrict__ stats) {
  __shared__ float si[512];  // 3x128 img channels + 128 mask
  const int HW = 65536;
  int p0 = blockIdx.x * 128;
  for (int i = threadIdx.x; i < 128; i += 192) {
    int pg = p0 + i;
    int b = pg / HW; int off = pg - b * HW;
    const float* ib = img + (size_t)b * 3 * HW + off;
    si[i]       = ib[0];
    si[128 + i] = ib[HW];
    si[256 + i] = ib[2 * HW];
    si[384 + i] = m0[pg];
  }
  __syncthreads();
  int o = threadIdx.x;
  float w0 = w[o * 3], w1 = w[o * 3 + 1], w2 = w[o * 3 + 2];
  float s = 0.f, sq = 0.f;
  for (int i = 0; i < 128; i++) {
    float m = si[384 + i];
    if (m != 0.f) {
      float v = w0 * si[i] + w1 * si[128 + i] + w2 * si[256 + i];
      s += v; sq += v * v;
    }
  }
  atomicAdd(&stats[o], s);
  atomicAdd(&stats[HD + o], sq);
}

__global__ void stem_apply(const float* __restrict__ img, const float* __restrict__ m0,
                           const float* __restrict__ w, const float* __restrict__ g,
                           const float* __restrict__ bb, const float* __restrict__ stats,
                           const float* __restrict__ cnt, _Float16* __restrict__ out, int B) {
  const int Hp = 258, HW = 65536;
  long long total = (long long)B * Hp * Hp * HD;
  long long idx = (long long)blockIdx.x * 256 + threadIdx.x;
  if (idx >= total) return;
  int o = (int)(idx % HD); long long t = idx / HD;
  int xp = (int)(t % Hp); t /= Hp; int yp = (int)(t % Hp); int b = (int)(t / Hp);
  float v = 0.f;
  if (xp > 0 && xp < 257 && yp > 0 && yp < 257) {
    int off = (yp - 1) * 256 + (xp - 1);
    float m = m0[b * HW + off];
    if (m != 0.f) {
      const float* ib = img + (size_t)b * 3 * HW + off;
      float conv = w[o * 3] * ib[0] + w[o * 3 + 1] * ib[HW] + w[o * 3 + 2] * ib[2 * HW];
      float c = fmaxf(cnt[0], 1.f);
      float mean = stats[o] / c;
      float var = stats[HD + o] / c - mean * mean;
      v = (conv - mean) * rsqrtf(var + EPSV) * g[o] + bb[o];
      v = fmaxf(v, 0.f);
    }
  }
  out[idx] = (_Float16)v;
}

// ---------------------------------------------------------------- weight packing
// packed[mt][kc][lane][slot] = exact A-fragment register image for
// v_wmma_f32_16x16x32_f16 (contiguous k order: k = 16*(lane/16) + slot), K = (ky*KW+kx)*HD + c
__global__ void pack_weights(const float* __restrict__ w, _Float16* __restrict__ packed,
                             int KW, int nk) {
  int idx = blockIdx.x * 256 + threadIdx.x;
  int total = 12 * nk * 512;
  if (idx >= total) return;
  int slot = idx & 15;
  int lane = (idx >> 4) & 31;
  int kc = (idx >> 9) % nk;
  int mt = idx / (nk * 512);
  int koff = slot;
  if (lane >= 16) koff += 16;
  int K = kc * 32 + koff;
  int m = mt * 16 + (lane & 15);
  int kxy = K / HD; int c = K - kxy * HD;
  int ky = kxy / KW; int kx = kxy - ky * KW;
  packed[idx] = (_Float16)w[(((size_t)m * HD + c) * KW + ky) * KW + kx];
}

// ---------------------------------------------------------------- implicit-GEMM conv
// One wave computes a 2x2 group of 16x16 tiles: 2 Cout tiles x 2 pixel tiles.
// K-loop is tap x channel-chunk nested (no divisions, unrolled inner chunk loop:
// 24 WMMAs per tap). A staged in LDS at tap granularity (5 taps / 60 KB phase).
#define WMMA16(A, Bv, C) __builtin_amdgcn_wmma_f32_16x16x32_f16(false, A, false, Bv, (short)0, C, false, false)

__global__ void __launch_bounds__(256)
conv_wmma(const _Float16* __restrict__ xin, const _Float16* __restrict__ packedA,
          float* __restrict__ out, int H, int W, int Hp, int Wp,
          int stride, int KW, int nk, int in_off,
          int tprShift, int tpbShift, int totalTiles) {
  __shared__ __align__(16) _Float16 ldsA[2 * TPP * 6 * 512];  // 60 KB
  const int tid = threadIdx.x;
  const int mt0 = blockIdx.y * 2;

  int lane = tid & 31, wave = tid >> 5;
  int t0 = (blockIdx.x * 8 + wave) * 2;

  int n = lane & 15;             // pixel within tile (N)
  int chalf = (lane >> 4) << 4;  // lanes 16..31 take K upper half

  // decode both pixel tiles (pure shifts); invalid tiles alias tile 0
  int tb[2], ty[2], tx[2]; bool val[2];
  for (int q = 0; q < 2; q++) {
    int t = t0 + q; val[q] = (t < totalTiles);
    int tt = val[q] ? t : 0;
    int b = tt >> tpbShift; int r = tt & ((1 << tpbShift) - 1);
    int y = r >> tprShift;  int x0 = (r & ((1 << tprShift) - 1)) << 4;
    tb[q] = b; ty[q] = y; tx[q] = x0;
  }

  v8f acc00 = {}, acc01 = {}, acc10 = {}, acc11 = {};

  const int tapsTotal = KW * KW;
  for (int tap0 = 0; tap0 < tapsTotal; tap0 += TPP) {
    int tcnt = tapsTotal - tap0; if (tcnt > TPP) tcnt = TPP;
    // stage this phase's A fragments (both Cout tiles) into LDS
    const uint4* s0 = (const uint4*)(packedA + ((size_t)mt0 * nk + tap0 * 6) * 512);
    const uint4* s1 = (const uint4*)(packedA + ((size_t)(mt0 + 1) * nk + tap0 * 6) * 512);
    uint4* d0 = (uint4*)ldsA;
    uint4* d1 = (uint4*)(ldsA + TPP * 6 * 512);
    int nvec = tcnt * 6 * 64;  // chunks * 64 uint4
    __syncthreads();           // protect previous phase's LDS reads
    for (int i = tid; i < nvec; i += 256) { d0[i] = s0[i]; d1[i] = s1[i]; }
    __syncthreads();

    for (int tl = 0; tl < tcnt; tl++) {
      int tap = tap0 + tl;
      int ky = tap / KW, kx = tap - ky * KW;  // tiny trip count, cheap
      const _Float16* base0 =
          xin + (((size_t)tb[0] * Hp + ty[0] * stride + ky + in_off) * Wp +
                 (tx[0] + n) * stride + kx + in_off) * HD + chalf;
      const _Float16* base1 =
          xin + (((size_t)tb[1] * Hp + ty[1] * stride + ky + in_off) * Wp +
                 (tx[1] + n) * stride + kx + in_off) * HD + chalf;
      const _Float16* la0 = ldsA + (size_t)(tl * 6) * 512 + lane * 16;
      const _Float16* la1 = ldsA + (size_t)(TPP * 6 + tl * 6) * 512 + lane * 16;

      // prefetch next tap's image rows (lowers to global_prefetch_b8)
      if (tl + 1 < tcnt) {
        int tn = tap + 1; int kyn = tn / KW, kxn = tn - kyn * KW;
        __builtin_prefetch(xin + (((size_t)tb[0] * Hp + ty[0] * stride + kyn + in_off) * Wp +
                                  (tx[0] + n) * stride + kxn + in_off) * HD + chalf, 0, 3);
        __builtin_prefetch(xin + (((size_t)tb[1] * Hp + ty[1] * stride + kyn + in_off) * Wp +
                                  (tx[1] + n) * stride + kxn + in_off) * HD + chalf, 0, 3);
      }

#pragma unroll
      for (int cc = 0; cc < 6; cc++) {
        v16h a0 = *(const v16h*)(la0 + cc * 512);
        v16h a1 = *(const v16h*)(la1 + cc * 512);
        BU b0, b1;
        b0.h[0] = *(const v8h*)(base0 + cc * 32);
        b0.h[1] = *(const v8h*)(base0 + cc * 32 + 8);
        b1.h[0] = *(const v8h*)(base1 + cc * 32);
        b1.h[1] = *(const v8h*)(base1 + cc * 32 + 8);
        acc00 = WMMA16(a0, b0.v, acc00);
        acc01 = WMMA16(a1, b0.v, acc01);
        acc10 = WMMA16(a0, b1.v, acc10);
        acc11 = WMMA16(a1, b1.v, acc11);
      }
    }
  }

  // C/D layout: lanes 0-15 -> M = r, lanes 16-31 -> M = 8 + r ; N = lane&15
  int rbase = (lane >> 4) << 3;
  for (int q = 0; q < 2; q++) {
    if (!val[q]) continue;
    float* opb = out + (((size_t)tb[q] * H + ty[q]) * W + (tx[q] + n)) * HD + rbase;
    v8f* accs0 = q ? &acc10 : &acc00;
    v8f* accs1 = q ? &acc11 : &acc01;
    float* op0 = opb + (size_t)mt0 * 16;
    float* op1 = opb + (size_t)(mt0 + 1) * 16;
    for (int rr = 0; rr < 8; rr++) {
      op0[rr] = (*accs0)[rr];
      op1[rr] = (*accs1)[rr];
    }
  }
}

// ---------------------------------------------------------------- masked BN
__global__ void __launch_bounds__(192)
bn_stats(const float* __restrict__ cbuf, const float* __restrict__ mask,
         float* __restrict__ stats, int npix) {
  __shared__ float ms[128];
  int p0 = blockIdx.x * 128;
  for (int i = threadIdx.x; i < 128; i += 192) {
    int p = p0 + i;
    ms[i] = (p < npix) ? mask[p] : 0.f;
  }
  __syncthreads();
  int o = threadIdx.x;
  float s = 0.f, sq = 0.f;
  int pend = (p0 + 128 < npix) ? p0 + 128 : npix;
  for (int p = p0; p < pend; p++) {
    float m = ms[p - p0];
    if (m != 0.f) {
      float v = cbuf[(size_t)p * HD + o];
      s += v; sq += v * v;
    }
  }
  atomicAdd(&stats[o], s);
  atomicAdd(&stats[HD + o], sq);
}

__global__ void apply_bn(const float* __restrict__ cbuf, const float* __restrict__ mask,
                         const float* __restrict__ stats, const float* __restrict__ cnt,
                         const float* __restrict__ g, const float* __restrict__ bb,
                         const _Float16* __restrict__ resid, _Float16* __restrict__ out,
                         int B, int H, int W, int relu) {
  int Hp = H + 2, Wp = W + 2;
  long long total = (long long)B * Hp * Wp * HD;
  long long idx = (long long)blockIdx.x * 256 + threadIdx.x;
  if (idx >= total) return;
  int o = (int)(idx % HD); long long t = idx / HD;
  int xp = (int)(t % Wp); t /= Wp; int yp = (int)(t % Hp); int b = (int)(t / Hp);
  float v = 0.f;
  if (xp > 0 && xp <= W && yp > 0 && yp <= H) {
    int p = (b * H + (yp - 1)) * W + (xp - 1);
    float m = mask[p];
    if (m != 0.f) {
      float c = fmaxf(cnt[0], 1.f);
      float mean = stats[o] / c;
      float var = stats[HD + o] / c - mean * mean;
      v = (cbuf[(size_t)p * HD + o] - mean) * rsqrtf(var + EPSV) * g[o] + bb[o];
      if (resid) v += (float)resid[idx];
      if (relu) v = fmaxf(v, 0.f);
    }
  }
  out[idx] = (_Float16)v;
}

__global__ void apply_final(const float* __restrict__ cbuf, const float* __restrict__ mask,
                            const float* __restrict__ stats, const float* __restrict__ cnt,
                            const float* __restrict__ g, const float* __restrict__ bb,
                            float* __restrict__ out, int B) {
  const int HW = 256;  // 16x16
  int total = B * HW * HD;
  int idx = blockIdx.x * 256 + threadIdx.x;
  if (idx >= total) return;
  int o = idx % HD; int p = idx / HD;
  int b = p / HW; int off = p - b * HW;
  float v = 0.f;
  float m = mask[p];
  if (m != 0.f) {
    float c = fmaxf(cnt[0], 1.f);
    float mean = stats[o] / c;
    float var = stats[HD + o] / c - mean * mean;
    v = (cbuf[(size_t)p * HD + o] - mean) * rsqrtf(var + EPSV) * g[o] + bb[o];
    v = fmaxf(v, 0.f);
  }
  out[((size_t)b * HD + o) * HW + off] = v;
}

__global__ void write_meta(const int* __restrict__ mask, float* __restrict__ out) {
  int i = blockIdx.x * 256 + threadIdx.x;
  if (i < 2048) {
    out[i] = (float)mask[i];          // mask, [B,256]
    out[2048 + i] = (float)(i & 255); // ids_restore, [B,256]
  }
}

// ---------------------------------------------------------------- host orchestration
extern "C" void kernel_launch(void* const* d_in, const int* in_sizes, int n_in,
                              void* d_out, int out_size, void* d_ws, size_t ws_size,
                              hipStream_t stream) {
  (void)in_sizes; (void)n_in; (void)out_size; (void)ws_size;
  const float* images = (const float*)d_in[0];
  const int*   maskI  = (const int*)d_in[1];
  const float* stem_w = (const float*)d_in[2];
  const float* stem_g = (const float*)d_in[3];
  const float* stem_b = (const float*)d_in[4];
  const float* dw1 = (const float*)d_in[5];
  const float* dg1 = (const float*)d_in[6];
  const float* db1 = (const float*)d_in[7];
  const float* dw2 = (const float*)d_in[8];
  const float* dg2 = (const float*)d_in[9];
  const float* db2 = (const float*)d_in[10];
  const float* rw1 = (const float*)d_in[11];
  const float* rg1 = (const float*)d_in[12];
  const float* rb1 = (const float*)d_in[13];
  const float* rw2 = (const float*)d_in[14];
  const float* rg2 = (const float*)d_in[15];
  const float* rb2 = (const float*)d_in[16];
  const float* fw  = (const float*)d_in[17];
  const float* fg  = (const float*)d_in[18];
  const float* fb  = (const float*)d_in[19];
  float* out = (float*)d_out;

  const int B = 8;
  char* p = (char*)d_ws;
  auto take = [&](size_t bytes) -> char* {
    char* q = p; p += (bytes + 255) & ~(size_t)255; return q;
  };
  _Float16* x0   = (_Float16*)take((size_t)B * 258 * 258 * HD * 2);
  _Float16* bufA = (_Float16*)take((size_t)B * 130 * 130 * HD * 2);
  _Float16* bufZ = (_Float16*)take((size_t)B * 130 * 130 * HD * 2);
  _Float16* bufX = (_Float16*)take((size_t)B * 130 * 130 * HD * 2);
  float*    cbuf = (float*)take((size_t)B * 128 * 128 * HD * 4);
  _Float16* packW = (_Float16*)take((size_t)54 * 12 * 512 * 2);
  int Hs[6] = {256, 128, 64, 32, 16, 8};
  float* mlev[5];
  for (int l = 0; l < 5; l++) mlev[l] = (float*)take((size_t)B * Hs[l] * Hs[l] * 4);
  float* stats = (float*)take(2 * HD * 4);
  float* cnts  = (float*)take(8 * 4);

  // masks + counts
  zero_f32<<<1, 256, 0, stream>>>(cnts, 8);
  make_mask0<<<(B * 65536) / 256, 256, 0, stream>>>(maskI, mlev[0], &cnts[0]);
  for (int l = 0; l < 4; l++) {
    int Ho = Hs[l + 1];
    int total = B * Ho * Ho;
    down_mask<<<(total + 255) / 256, 256, 0, stream>>>(mlev[l], mlev[l + 1],
                                                       &cnts[l + 1], B, Hs[l], Ho);
  }

  // stem: stats then recompute+apply, f16 halo-padded output
  zero_f32<<<2, 256, 0, stream>>>(stats, 2 * HD);
  stem_stats<<<(B * 65536) / 128, 192, 0, stream>>>(images, mlev[0], stem_w, stats);
  {
    long long tot = (long long)B * 258 * 258 * HD;
    stem_apply<<<(int)((tot + 255) / 256), 256, 0, stream>>>(
        images, mlev[0], stem_w, stem_g, stem_b, stats, &cnts[0], x0, B);
  }

  auto log2i = [](int v) { int s = 0; while ((1 << s) < v) s++; return s; };

  auto run_conv = [&](const float* w, const _Float16* xin, int Hp, int H,
                      int stride, int KW) {
    int nk = (KW == 3) ? 54 : 6;
    pack_weights<<<(12 * nk * 512 + 255) / 256, 256, 0, stream>>>(w, packW, KW, nk);
    int tpr = H / 16;                  // tiles per row (W == H)
    int totalTiles = B * H * tpr;
    dim3 g((totalTiles + 15) / 16, 6);  // 8 waves x 2 pixel tiles per WG; 6 Cout pairs
    conv_wmma<<<g, 256, 0, stream>>>(xin, packW, cbuf, H, H, Hp, Hp,
                                     stride, KW, nk, (KW == 1) ? 1 : 0,
                                     log2i(tpr), log2i(H * tpr), totalTiles);
  };
  auto run_bn = [&](int lvl, int H, const float* g, const float* b,
                    _Float16* outB, const _Float16* resid, int relu) {
    int npix = B * H * H;
    zero_f32<<<2, 256, 0, stream>>>(stats, 2 * HD);
    bn_stats<<<(npix + 127) / 128, 192, 0, stream>>>(cbuf, mlev[lvl], stats, npix);
    long long tot = (long long)B * (H + 2) * (H + 2) * HD;
    apply_bn<<<(int)((tot + 255) / 256), 256, 0, stream>>>(
        cbuf, mlev[lvl], stats, &cnts[lvl], g, b, resid, outB, B, H, H, relu);
  };

  const _Float16* xcur = x0;
  int HpCur = 258;
  const size_t w9 = (size_t)HD * HD * 9;
  for (int i = 0; i < 4; i++) {
    int H = Hs[i + 1];
    // ResBlockDown: conv s2 -> BN+ReLU -> subm conv -> BN
    run_conv(dw1 + i * w9, xcur, HpCur, H, 2, 3);
    run_bn(i + 1, H, dg1 + i * HD, db1 + i * HD, bufA, nullptr, 1);
    run_conv(dw2 + i * w9, bufA, H + 2, H, 1, 3);
    run_bn(i + 1, H, dg2 + i * HD, db2 + i * HD, bufZ, nullptr, 0);
    // ResBlock: conv -> BN+ReLU -> conv -> BN, +identity, ReLU
    run_conv(rw1 + i * w9, bufZ, H + 2, H, 1, 3);
    run_bn(i + 1, H, rg1 + i * HD, rb1 + i * HD, bufA, nullptr, 1);
    run_conv(rw2 + i * w9, bufA, H + 2, H, 1, 3);
    run_bn(i + 1, H, rg2 + i * HD, rb2 + i * HD, bufX, bufZ, 1);
    xcur = bufX; HpCur = H + 2;
  }

  // final 1x1 proj + BN + ReLU -> f32 NCHW into d_out
  run_conv(fw, bufX, 18, 16, 1, 1);
  {
    int npix = B * 256;
    zero_f32<<<2, 256, 0, stream>>>(stats, 2 * HD);
    bn_stats<<<(npix + 127) / 128, 192, 0, stream>>>(cbuf, mlev[4], stats, npix);
    int tot = B * 256 * HD;
    apply_final<<<(tot + 255) / 256, 256, 0, stream>>>(cbuf, mlev[4], stats,
                                                       &cnts[4], fg, fb, out, B);
  }
  write_meta<<<8, 256, 0, stream>>>(maskI, out + (size_t)B * HD * 256);
}
